// myModel_16999480558113
// MI455X (gfx1250) — matrix operand order repaired and checked by reading, not compiled, not run on hardware
//
#include <hip/hip_runtime.h>

typedef __bf16 v16bf __attribute__((ext_vector_type(16)));
typedef float  v8f   __attribute__((ext_vector_type(8)));

#define ALPHA 0.2f
#define LDW 20  // padded row stride (dwords) for packed K-pair LDS tiles: 80B, 16B-aligned

static __device__ __forceinline__ float lrelu(float x) { return x > 0.f ? x : ALPHA * x; }

static __device__ __forceinline__ uint pack_bf16(float lo, float hi) {
  union { __bf16 h[2]; uint u; } w;
  w.h[0] = (__bf16)lo;
  w.h[1] = (__bf16)hi;
  return w.u;
}

// ============== CDNA5 async global->LDS copies (ASYNCcnt-tracked, no VGPR bounce) =========
static __device__ __forceinline__ void async_copy_b128(void* lds, const void* g) {
  asm volatile("global_load_async_to_lds_b128 %0, %1, off"
               :: "v"((uint)(uintptr_t)lds), "v"((unsigned long long)(uintptr_t)g)
               : "memory");
}
static __device__ __forceinline__ void async_copy_b64(void* lds, const void* g) {
  asm volatile("global_load_async_to_lds_b64 %0, %1, off"
               :: "v"((uint)(uintptr_t)lds), "v"((unsigned long long)(uintptr_t)g)
               : "memory");
}
static __device__ __forceinline__ void wait_async_all() {
  asm volatile("s_wait_asynccnt 0" ::: "memory");
}

// ================= packed-pair fragment loaders (wave32 16x16x32 bf16 WMMA) ==============
// LDS word = {bf16 K=2k (lo), bf16 K=2k+1 (hi)}.
static __device__ __forceinline__ v16bf load_frag_a_packed(const uint* As, int row0) {
  int lane = threadIdx.x & 31;
  int m = (lane & 15) + row0;
  int ho = (lane >> 4) << 2;
  union { uint4 q[2]; v16bf v; } u;
  u.q[0] = *(const uint4*)(As + m * LDW + ho);
  u.q[1] = *(const uint4*)(As + m * LDW + 8 + ho);
  return u.v;
}

static __device__ __forceinline__ v16bf load_frag_b_packed(const uint* Bst, int ncol) {
  int lane = threadIdx.x & 31;
  int n = (lane & 15) + ncol;
  int ko2 = (lane >> 4) << 2;
  const uint4* p = (const uint4*)(Bst + n * LDW + ko2);
  union { uint4 q[2]; v16bf v; } u;
  u.q[0] = p[0];
  u.q[1] = p[2];
  return u.v;
}

// Stage one 32-K chunk of a pre-packed N-major matrix (row stride Kp dwords) into LDS,
// using async global->LDS b128 (2 per thread, 256 threads cover 128 rows x 16 kpairs).
static __device__ __forceinline__ void stage_b_async(uint* Bst, const uint* __restrict__ Btp,
                                                     int kc, int Kp) {
  int t = threadIdx.x;
  int n = t >> 1;
  int hw = (t & 1) << 3;  // 0 or 8 kpairs
  const uint* g = Btp + (size_t)n * Kp + (kc >> 1) + hw;
  async_copy_b128(&Bst[n * LDW + hw], g);
  async_copy_b128(&Bst[n * LDW + hw + 4], g + 4);
}

// Stage 32x32 A chunk (row-major bf16, row stride K elems) -> packed LDS, async b64/thread.
static __device__ __forceinline__ void stage_a_async(uint* As, const __bf16* __restrict__ Ap,
                                                     int kc, int K) {
  int t = threadIdx.x;
  int w = t * 2;
  int r = w >> 4, kp = w & 15;  // kp even
  async_copy_b64(&As[r * LDW + kp], Ap + (size_t)r * K + kc + 2 * kp);
}

// ================= fp32 -> bf16 convert =================
__global__ void cvt_bf16_kernel(const float* __restrict__ in, __bf16* __restrict__ out, int n) {
  int i = blockIdx.x * blockDim.x + threadIdx.x;
  if (i < n) out[i] = (__bf16)in[i];
}

// ======= pack weights: f32 W[z][K][128] -> uint Wt[z][128][K/2] (K-pair, N-major) =======
__global__ void pack_wt_kernel(const float* __restrict__ W, uint* __restrict__ Wt,
                               int kpBits, long strideWz) {
  size_t idx = (size_t)blockIdx.x * 256 + threadIdx.x;
  int Kp = 1 << kpBits;
  int kp = (int)(idx & (Kp - 1));
  size_t zn = idx >> kpBits;
  int n = (int)(zn & 127);
  int z = (int)(zn >> 7);
  const float* Wp = W + (size_t)z * strideWz;
  Wt[idx] = pack_bf16(Wp[(size_t)(2 * kp) * 128 + n], Wp[(size_t)(2 * kp + 1) * 128 + n]);
}

// ===== bf16 GEMM: C[z][M x 128] = A[M x K] @ B[z][K x 128], B pre-packed N-major =====
// 32x128 tile, 8 waves, double-buffered async pipeline: stage chunk c+1 while computing c.
__global__ void gemm_bf16_kernel(const __bf16* __restrict__ A, const uint* __restrict__ Bt,
                                 __bf16* __restrict__ C, uint* __restrict__ Ct, int K,
                                 long strideBtz, long strideCz) {
  __shared__ uint As[2][32 * LDW];
  __shared__ uint Bst[2][128 * LDW];
  int rt = blockIdx.x, z = blockIdx.z;
  int Kp = K >> 1;
  const __bf16* Ap = A + (size_t)rt * 32 * K;
  const uint* Btp = Bt + (size_t)z * strideBtz;
  __bf16* Cp = C + (size_t)z * strideCz + (size_t)rt * 32 * 128;
  int t = threadIdx.x, wave = t >> 5, lane = t & 31;
  v8f acc0 = {}, acc1 = {};

  stage_a_async(As[0], Ap, 0, K);
  stage_b_async(Bst[0], Btp, 0, Kp);
  wait_async_all();
  __syncthreads();

  int nchunks = K >> 5;
  for (int c = 0; c < nchunks; ++c) {
    int cur = c & 1, nxt = cur ^ 1;
    if (c + 1 < nchunks) {
      stage_a_async(As[nxt], Ap, (c + 1) * 32, K);
      stage_b_async(Bst[nxt], Btp, (c + 1) * 32, Kp);
    }
    v16bf bfg = load_frag_b_packed(Bst[cur], wave * 16);
    v16bf a0 = load_frag_a_packed(As[cur], 0);
    v16bf a1 = load_frag_a_packed(As[cur], 16);
    acc0 = __builtin_amdgcn_wmma_f32_16x16x32_bf16(false, a0, false, bfg, (short)0, acc0,
                                                   false, false);
    acc1 = __builtin_amdgcn_wmma_f32_16x16x32_bf16(false, a1, false, bfg, (short)0, acc1,
                                                   false, false);
    wait_async_all();
    __syncthreads();
  }

  int n = lane & 15;
  int mo = (lane >> 4) * 8;
  int col = wave * 16 + n;
#pragma unroll
  for (int r = 0; r < 8; ++r) {
    Cp[(size_t)(mo + r) * 128 + col] = (__bf16)acc0[r];
    Cp[(size_t)(16 + mo + r) * 128 + col] = (__bf16)acc1[r];
  }
  // packed-transposed: Ct[(zb*128 + col)*512 + jpair], jpair = global_row/2 within b-block
  int b_idx = (rt * 32) >> 10;
  int jb = (((rt * 32) & 1023) >> 1) + (mo >> 1);
  uint* ht = Ct + ((size_t)(z * 8 + b_idx) * 128 + col) * 512 + jb;
  uint4 q0, q1;
  q0.x = pack_bf16(acc0[0], acc0[1]); q0.y = pack_bf16(acc0[2], acc0[3]);
  q0.z = pack_bf16(acc0[4], acc0[5]); q0.w = pack_bf16(acc0[6], acc0[7]);
  q1.x = pack_bf16(acc1[0], acc1[1]); q1.y = pack_bf16(acc1[2], acc1[3]);
  q1.z = pack_bf16(acc1[4], acc1[5]); q1.w = pack_bf16(acc1[6], acc1[7]);
  *(uint4*)ht = q0;
  *(uint4*)(ht + 8) = q1;  // acc1 rows are +16 -> +8 jpairs
}

// ================= f1 = h@a1, f2 = h@a2 (one wave per row, F=128) =================
__global__ void row_f12_kernel(const __bf16* __restrict__ hb, const float* __restrict__ a,
                               float* __restrict__ f1, float* __restrict__ f2, int rowsPerZ) {
  int wave = threadIdx.x >> 5, lane = threadIdx.x & 31;
  size_t row = (size_t)blockIdx.x * 8 + wave;
  int z = (int)(row / rowsPerZ);
  const __bf16* h = hb + row * 128;
  const float* a1 = a + (size_t)z * 256;
  const float* a2 = a1 + 128;
  float s1 = 0.f, s2 = 0.f;
#pragma unroll
  for (int q = 0; q < 4; ++q) {
    float hv = (float)h[lane + 32 * q];
    s1 += hv * a1[lane + 32 * q];
    s2 += hv * a2[lane + 32 * q];
  }
#pragma unroll
  for (int off = 16; off >= 1; off >>= 1) {
    s1 += __shfl_xor(s1, off, 32);
    s2 += __shfl_xor(s2, off, 32);
  }
  if (lane == 0) { f1[row] = s1; f2[row] = s2; }
}

// ====== max_j f2 per (z,b) (leakyrelu monotone => row max = lrelu(f1_i + maxf2)) ======
__global__ void colmax_kernel(const float* __restrict__ f2, float* __restrict__ mx) {
  __shared__ float red[256];
  const float* p = f2 + (size_t)blockIdx.x * 1024;
  int t = threadIdx.x;
  float m = fmaxf(fmaxf(p[t], p[t + 256]), fmaxf(p[t + 512], p[t + 768]));
  red[t] = m;
  __syncthreads();
  for (int s = 128; s > 0; s >>= 1) {
    if (t < s) red[t] = fmaxf(red[t], red[t + s]);
    __syncthreads();
  }
  if (t == 0) mx[blockIdx.x] = red[0];
}

// ================= row sum of exp(lrelu(f1_i+f2_j) - m_i) =================
__global__ void sumexp_kernel(const float* __restrict__ f1, const float* __restrict__ f2,
                              const float* __restrict__ mx, float* __restrict__ ssum) {
  int wave = threadIdx.x >> 5, lane = threadIdx.x & 31;
  size_t row = (size_t)blockIdx.x * 8 + wave;
  int zb = (int)(row >> 10);
  float f1v = f1[row];
  float m = lrelu(f1v + mx[zb]);
  const float* f2p = f2 + ((size_t)zb << 10);
  float s = 0.f;
  for (int j = lane; j < 1024; j += 32) s += __expf(lrelu(f1v + f2p[j]) - m);
#pragma unroll
  for (int off = 16; off >= 1; off >>= 1) s += __shfl_xor(s, off, 32);
  if (lane == 0) ssum[row] = s;
}

// ======== fused softmax(P) @ H via WMMA; attention matrix never materialized ========
// grid: (N/32, B, heads). H pre-packed/transposed by GEMM epilogue; async + double buffer.
template <bool FINAL>
__global__ void attn_av_kernel(const uint* __restrict__ Ht, const float* __restrict__ f1,
                               const float* __restrict__ f2, const float* __restrict__ mx,
                               const float* __restrict__ ssum, void* __restrict__ outp) {
  __shared__ uint Ps[2][32 * LDW];
  __shared__ uint Hst[2][128 * LDW];
  __shared__ float rf1[32], rm[32], rinv[32];
  int rt = blockIdx.x, b = blockIdx.y, z = blockIdx.z;
  int zb = z * 8 + b;
  size_t rowbase = (size_t)zb * 1024;
  const uint* Htp = Ht + (size_t)zb * (128 * 512);
  const float* f2row = f2 + rowbase;
  int t = threadIdx.x, wave = t >> 5, lane = t & 31;
  if (t < 32) {
    float fv = f1[rowbase + rt * 32 + t];
    rf1[t] = fv;
    rm[t] = lrelu(fv + mx[zb]);
    rinv[t] = 1.f / ssum[rowbase + rt * 32 + t];
  }
  __syncthreads();

  // P-tile builder: 32x32 softmax probabilities -> packed K-pairs (uint2 per thread)
  int pr = (t * 2) >> 4, pkp = (t * 2) & 15;  // row, even kpair
  auto stage_p = [&](uint* Pbuf, int kc) {
    float fr = rf1[pr], mr = rm[pr], ir = rinv[pr];
    int jb = kc + 2 * pkp;
    float p0 = __expf(lrelu(fr + f2row[jb + 0]) - mr) * ir;
    float p1 = __expf(lrelu(fr + f2row[jb + 1]) - mr) * ir;
    float p2 = __expf(lrelu(fr + f2row[jb + 2]) - mr) * ir;
    float p3 = __expf(lrelu(fr + f2row[jb + 3]) - mr) * ir;
    uint2 pw;
    pw.x = pack_bf16(p0, p1);
    pw.y = pack_bf16(p2, p3);
    *(uint2*)&Pbuf[pr * LDW + pkp] = pw;
  };

  v8f acc0 = {}, acc1 = {};
  stage_p(Ps[0], 0);
  stage_b_async(Hst[0], Htp, 0, 512);
  wait_async_all();
  __syncthreads();

  for (int c = 0; c < 32; ++c) {
    int cur = c & 1, nxt = cur ^ 1;
    if (c < 31) {
      stage_p(Ps[nxt], (c + 1) * 32);
      stage_b_async(Hst[nxt], Htp, (c + 1) * 32, 512);
    }
    v16bf bfg = load_frag_b_packed(Hst[cur], wave * 16);
    v16bf a0 = load_frag_a_packed(Ps[cur], 0);
    v16bf a1 = load_frag_a_packed(Ps[cur], 16);
    acc0 = __builtin_amdgcn_wmma_f32_16x16x32_bf16(false, a0, false, bfg, (short)0, acc0,
                                                   false, false);
    acc1 = __builtin_amdgcn_wmma_f32_16x16x32_bf16(false, a1, false, bfg, (short)0, acc1,
                                                   false, false);
    wait_async_all();
    __syncthreads();
  }

  int n = lane & 15;
  int mo = (lane >> 4) * 8;
#pragma unroll
  for (int half = 0; half < 2; ++half) {
    v8f& acc = half ? acc1 : acc0;
#pragma unroll
    for (int r = 0; r < 8; ++r) {
      int row = rt * 32 + half * 16 + mo + r;
      float v = acc[r];
      v = v > 0.f ? v : (__expf(v) - 1.f);  // elu inside _gat_layer
      if (FINAL) {
        v = v > 0.f ? v : (__expf(v) - 1.f);  // outer elu in reference
        ((float*)outp)[((size_t)b * 1024 + row) * 128 + wave * 16 + n] = v;
      } else {
        // concat layout: xc[b][n][z*128 + f], bf16 (A operand of layer-2 GEMM)
        ((__bf16*)outp)[((size_t)b * 1024 + row) * 1024 + (size_t)z * 128 + wave * 16 + n] =
            (__bf16)v;
      }
    }
  }
}

extern "C" void kernel_launch(void* const* d_in, const int* in_sizes, int n_in,
                              void* d_out, int out_size, void* d_ws, size_t ws_size,
                              hipStream_t stream) {
  const float* x  = (const float*)d_in[0];
  // d_in[1] = adj: unused (reference discards the mask)
  const float* Wh = (const float*)d_in[2];
  const float* ah = (const float*)d_in[3];
  const float* Wo = (const float*)d_in[4];
  const float* ao = (const float*)d_in[5];

  char* ws = (char*)d_ws;
  size_t off = 0;
  auto take = [&](size_t bytes) -> char* {
    char* p = ws + off;
    off += (bytes + 255) & ~(size_t)255;
    return p;
  };

  const int B = 8, N = 1024, D = 256, H = 8, F = 128;
  const int M = B * N;  // 8192 rows

  __bf16* xb  = (__bf16*)take((size_t)M * D * 2);               // x bf16 (GEMM-1 A)
  uint*   Wht = (uint*)take((size_t)H * 128 * (D / 2) * 4);     // packed W_heads
  uint*   Wot = (uint*)take((size_t)128 * (H * F / 2) * 4);     // packed W_out
  __bf16* hb  = (__bf16*)take((size_t)H * M * F * 2);           // layer-1 h row-major
  uint*   h1t = (uint*)take((size_t)H * B * 128 * 512 * 4);     // layer-1 h packed/T
  float*  f1a = (float*)take((size_t)H * M * 4);
  float*  f2a = (float*)take((size_t)H * M * 4);
  float*  s1a = (float*)take((size_t)H * M * 4);
  float*  mx1 = (float*)take((size_t)H * B * 4);
  __bf16* xc  = (__bf16*)take((size_t)M * (H * F) * 2);         // concat heads (GEMM-2 A)
  __bf16* h2b = (__bf16*)take((size_t)M * F * 2);               // layer-2 h row-major
  uint*   h2t = (uint*)take((size_t)B * 128 * 512 * 4);         // layer-2 h packed/T
  float*  f1b = (float*)take((size_t)M * 4);
  float*  f2b = (float*)take((size_t)M * 4);
  float*  s2b = (float*)take((size_t)M * 4);
  float*  mx2 = (float*)take((size_t)B * 4);
  (void)ws_size; (void)in_sizes; (void)n_in; (void)out_size;

  // ---- converts / packs ----
  cvt_bf16_kernel<<<(M * D + 255) / 256, 256, 0, stream>>>(x, xb, M * D);
  pack_wt_kernel<<<(H * 128 * (D / 2)) / 256, 256, 0, stream>>>(Wh, Wht, 7, (long)D * F);
  pack_wt_kernel<<<(128 * (H * F / 2)) / 256, 256, 0, stream>>>(Wo, Wot, 9, (long)H * F * F);

  // ---- layer 1: h = x @ W per head ----
  gemm_bf16_kernel<<<dim3(M / 32, 1, H), 256, 0, stream>>>(xb, Wht, hb, h1t, D,
                                                           (long)128 * (D / 2),
                                                           (long)M * F);
  row_f12_kernel<<<(H * M) / 8, 256, 0, stream>>>(hb, ah, f1a, f2a, M);
  colmax_kernel<<<H * B, 256, 0, stream>>>(f2a, mx1);
  sumexp_kernel<<<(H * M) / 8, 256, 0, stream>>>(f1a, f2a, mx1, s1a);
  attn_av_kernel<false><<<dim3(N / 32, B, H), 256, 0, stream>>>(h1t, f1a, f2a, mx1, s1a, xc);

  // ---- layer 2: h2 = xc @ W_out ----
  gemm_bf16_kernel<<<dim3(M / 32, 1, 1), 256, 0, stream>>>(xc, Wot, h2b, h2t, H * F, 0L, 0L);
  row_f12_kernel<<<M / 8, 256, 0, stream>>>(h2b, ao, f1b, f2b, M);
  colmax_kernel<<<B, 256, 0, stream>>>(f2b, mx2);
  sumexp_kernel<<<M / 8, 256, 0, stream>>>(f1b, f2b, mx2, s2b);
  attn_av_kernel<true><<<dim3(N / 32, B, 1), 256, 0, stream>>>(h2t, f1b, f2b, mx2, s2b,
                                                               d_out);
}
